// SelfAttentionHead_29643864277326
// MI455X (gfx1250) — matrix operand order repaired and checked by reading, not compiled, not run on hardware
//
#include <hip/hip_runtime.h>
#include <stdint.h>

typedef _Float16 v16h __attribute__((ext_vector_type(16)));
typedef _Float16 v8h  __attribute__((ext_vector_type(8)));
typedef float    v8f  __attribute__((ext_vector_type(8)));
typedef float    v4f  __attribute__((ext_vector_type(4)));

#define BM   128
#define BN   128
#define BK   32
#define PADH 8
#define LDT  (BK + PADH)   // 40 halves = 80B row stride -> conflict-free b128 LDS reads

static __device__ __forceinline__ v16h h16cat(v8h lo, v8h hi) {
  return __builtin_shufflevector(lo, hi, 0,1,2,3,4,5,6,7,8,9,10,11,12,13,14,15);
}

// A fragment (16x32 f16): lane&15 = M row; lanes>=16 take K chunks +8 / +24.
static __device__ __forceinline__ v16h load_a_frag(const _Float16* As, int row, int lane) {
  int r = row + (lane & 15);
  int k = (lane >> 4) * 8;
  v8h lo = *(const v8h*)(As + r * LDT + k);
  v8h hi = *(const v8h*)(As + r * LDT + 16 + k);
  return h16cat(lo, hi);
}

// B fragment (32x16 f16) from transposed tile Bt[n][k]: lane&15 = N col,
// lanes>=16 take K chunks +8 / +24 (same order as A).
static __device__ __forceinline__ v16h load_b_frag(const _Float16* Bt, int col, int lane) {
  int n = col + (lane & 15);
  int k = (lane >> 4) * 8;
  v8h b0 = *(const v8h*)(Bt + n * LDT + k);
  v8h b1 = *(const v8h*)(Bt + n * LDT + 16 + k);
  return h16cat(b0, b1);
}

static __device__ __forceinline__ v8h cvt8(v4f a, v4f b) {
  v8h r;
  r[0]=(_Float16)a[0]; r[1]=(_Float16)a[1]; r[2]=(_Float16)a[2]; r[3]=(_Float16)a[3];
  r[4]=(_Float16)b[0]; r[5]=(_Float16)b[1]; r[6]=(_Float16)b[2]; r[7]=(_Float16)b[3];
  return r;
}

// ---- CDNA5 async global->LDS copy (ASYNCcnt-tracked, no VGPR staging) -------------
static __device__ __forceinline__ uint32_t lds_off(const void* p) {
  return (uint32_t)(uintptr_t)p;   // generic LDS pointer: low 32 bits = LDS offset
}
static __device__ __forceinline__ void async_copy_b128(uint32_t ldsoff, const void* gptr) {
  asm volatile("global_load_async_to_lds_b128 %0, %1, off"
               :: "v"(ldsoff), "v"((unsigned long long)(uintptr_t)gptr)
               : "memory");
}
static __device__ __forceinline__ void wait_async0() {
#if __has_builtin(__builtin_amdgcn_s_wait_asynccnt)
  __builtin_amdgcn_s_wait_asynccnt(0);
#else
  asm volatile("s_wait_asynccnt 0x0" ::: "memory");
#endif
}

// 8 WMMAs per wave per k-step: 4 M-fragments x 2 N-fragments.
static __device__ __forceinline__ void mma_step(const _Float16* As, const _Float16* Bt,
                                                int wm, int wn, int lane, v8f acc[4][2]) {
  v16h a[4], b[2];
  #pragma unroll
  for (int fm = 0; fm < 4; ++fm) a[fm] = load_a_frag(As, wm + fm * 16, lane);
  #pragma unroll
  for (int fn = 0; fn < 2; ++fn) b[fn] = load_b_frag(Bt, wn + fn * 16, lane);
  #pragma unroll
  for (int fm = 0; fm < 4; ++fm)
    #pragma unroll
    for (int fn = 0; fn < 2; ++fn)
      acc[fm][fn] = __builtin_amdgcn_wmma_f32_16x16x32_f16(
          false, a[fm], false, b[fn], (short)0, acc[fm][fn], false, false);
}

// ---------------- Kernel 1: Out_f16 = X_f32 @ W_f32  (K=2048, N=2048) ---------------
__global__ __launch_bounds__(256) void gemm_xw_kernel(
    const float* __restrict__ X, const float* __restrict__ W,
    _Float16* __restrict__ Out, int Kdim, int Ndim)
{
  __shared__ _Float16 As[2][BM * LDT];
  __shared__ _Float16 Bt[2][BN * LDT];
  const int t    = threadIdx.x;
  const int lane = t & 31;
  const int wid  = t >> 5;
  const int r0   = blockIdx.y * BM;
  const int c0   = blockIdx.x * BN;
  const int wm   = (wid >> 2) * 64;
  const int wn   = (wid & 3) * 32;

  const int ar = t >> 1, ak = (t & 1) * 16;   // 16 elems/thread for A (128x32)
  const int bk = t >> 3, bn = (t & 7) * 16;   // 16 elems/thread for B (32x128)

  const float* pa = X + (size_t)(r0 + ar) * Kdim + ak;
  const float* pb = W + (size_t)bk * Ndim + c0 + bn;
  const size_t bstep = (size_t)BK * Ndim;

  v4f xa[4], xb[4];
  #pragma unroll
  for (int i = 0; i < 4; ++i) { xa[i] = *(const v4f*)(pa + 4*i); xb[i] = *(const v4f*)(pb + 4*i); }
  { // stage tile 0 into buffer 0
    _Float16* da = &As[0][ar * LDT + ak];
    *(v8h*)da       = cvt8(xa[0], xa[1]);
    *(v8h*)(da + 8) = cvt8(xa[2], xa[3]);
    #pragma unroll
    for (int q = 0; q < 4; ++q)
      #pragma unroll
      for (int j = 0; j < 4; ++j)
        Bt[0][(bn + q*4 + j) * LDT + bk] = (_Float16)xb[q][j];
  }

  v8f acc[4][2] = {};
  int cur = 0;
  for (int k0 = 0; k0 < Kdim; k0 += BK) {
    __syncthreads();
    const bool more = (k0 + BK) < Kdim;
    if (more) {
      pa += BK; pb += bstep;
      #pragma unroll
      for (int i = 0; i < 4; ++i) { xa[i] = *(const v4f*)(pa + 4*i); xb[i] = *(const v4f*)(pb + 4*i); }
      if ((k0 + 2*BK) < Kdim) {               // prefetch tile-after-next
        __builtin_prefetch(pa + BK, 0, 1);
        __builtin_prefetch(pb + bstep, 0, 1);
      }
    }
    mma_step(As[cur], Bt[cur], wm, wn, lane, acc);   // overlaps with global loads
    if (more) {
      const int nxt = cur ^ 1;
      _Float16* da = &As[nxt][ar * LDT + ak];
      *(v8h*)da       = cvt8(xa[0], xa[1]);
      *(v8h*)(da + 8) = cvt8(xa[2], xa[3]);
      #pragma unroll
      for (int q = 0; q < 4; ++q)
        #pragma unroll
        for (int j = 0; j < 4; ++j)
          Bt[nxt][(bn + q*4 + j) * LDT + bk] = (_Float16)xb[q][j];
    }
    cur ^= 1;
  }

  #pragma unroll
  for (int fm = 0; fm < 4; ++fm)
    #pragma unroll
    for (int fn = 0; fn < 2; ++fn) {
      int n  = c0 + wn + fn * 16 + (lane & 15);
      int mb = r0 + wm + fm * 16 + (lane >> 4) * 8;
      #pragma unroll
      for (int i = 0; i < 8; ++i)
        Out[(size_t)(mb + i) * Ndim + n] = (_Float16)acc[fm][fn][i];
    }
}

// ---------------- Kernel 2: S_f16 = Q16 @ K16^T (4096x4096, K=2048), causal skip ----
// A and B tiles move entirely via global_load_async_to_lds_b128 (no VGPR staging).
__global__ __launch_bounds__(256) void gemm_qk_kernel(
    const _Float16* __restrict__ Q, const _Float16* __restrict__ Kmat,
    _Float16* __restrict__ S, int Kdim, int Ndim)
{
  const int r0 = blockIdx.y * BM;
  const int c0 = blockIdx.x * BN;
  if (c0 > r0 + BM - 1) return;   // tile entirely above the diagonal: never read

  __shared__ _Float16 As[2][BM * LDT];
  __shared__ _Float16 Bt[2][BN * LDT];
  const int t    = threadIdx.x;
  const int lane = t & 31;
  const int wid  = t >> 5;
  const int wm   = (wid >> 2) * 64;
  const int wn   = (wid & 3) * 32;

  const int ar = t >> 1, ak = (t & 1) * 16;   // A: 128 rows x 32 k, 16 halves/thread
  const int bn = t >> 1, bk = (t & 1) * 16;   // B^T: contiguous reads of K rows

  const _Float16* pa = Q    + (size_t)(r0 + ar) * Kdim + ak;
  const _Float16* pb = Kmat + (size_t)(c0 + bn) * Kdim + bk;

  const uint32_t aoff[2] = { lds_off(&As[0][ar * LDT + ak]), lds_off(&As[1][ar * LDT + ak]) };
  const uint32_t boff[2] = { lds_off(&Bt[0][bn * LDT + bk]), lds_off(&Bt[1][bn * LDT + bk]) };

  async_copy_b128(aoff[0],      pa);
  async_copy_b128(aoff[0] + 16, pa + 8);
  async_copy_b128(boff[0],      pb);
  async_copy_b128(boff[0] + 16, pb + 8);

  v8f acc[4][2] = {};
  int cur = 0;
  for (int k0 = 0; k0 < Kdim; k0 += BK) {
    wait_async0();            // this wave's async LDS writes are done
    __syncthreads();          // everyone's writes visible; other buffer free
    const bool more = (k0 + BK) < Kdim;
    if (more) {
      pa += BK; pb += BK;
      const int nxt = cur ^ 1;
      async_copy_b128(aoff[nxt],      pa);
      async_copy_b128(aoff[nxt] + 16, pa + 8);
      async_copy_b128(boff[nxt],      pb);
      async_copy_b128(boff[nxt] + 16, pb + 8);
      if ((k0 + 2*BK) < Kdim) {
        __builtin_prefetch(pa + BK, 0, 1);
        __builtin_prefetch(pb + BK, 0, 1);
      }
    }
    mma_step(As[cur], Bt[cur], wm, wn, lane, acc);   // overlaps with async DMA
    cur ^= 1;
  }

  #pragma unroll
  for (int fm = 0; fm < 4; ++fm)
    #pragma unroll
    for (int fn = 0; fn < 2; ++fn) {
      int n  = c0 + wn + fn * 16 + (lane & 15);
      int mb = r0 + wm + fm * 16 + (lane >> 4) * 8;
      #pragma unroll
      for (int i = 0; i < 8; ++i)
        S[(size_t)(mb + i) * Ndim + n] = (_Float16)acc[fm][fn][i];
    }
}

// ---------------- Kernel 3: in-place causal softmax on S (rows of 4096) ------------
__global__ __launch_bounds__(256) void softmax_kernel(_Float16* __restrict__ S)
{
  const int   r      = blockIdx.x;
  const int   t      = threadIdx.x;
  const int   nvalid = r + 1;
  const float scale  = 0.022097086912079608f;  // 1/sqrt(2048)
  _Float16* row = S + (size_t)r * 4096;
  __shared__ float red[256];

  float m = -3.4e38f;
  for (int c = t; c < nvalid; c += 256)
    m = fmaxf(m, (float)row[c] * scale);
  red[t] = m; __syncthreads();
  for (int s = 128; s > 0; s >>= 1) { if (t < s) red[t] = fmaxf(red[t], red[t + s]); __syncthreads(); }
  const float M = red[0]; __syncthreads();

  float l = 0.0f;
  for (int c = t; c < nvalid; c += 256)
    l += __expf((float)row[c] * scale - M);
  red[t] = l; __syncthreads();
  for (int s = 128; s > 0; s >>= 1) { if (t < s) red[t] += red[t + s]; __syncthreads(); }
  const float invL = 1.0f / red[0]; __syncthreads();

  for (int c = t; c < 4096; c += 256) {
    float p = 0.0f;
    if (c < nvalid) p = __expf((float)row[c] * scale - M) * invL;
    row[c] = (_Float16)p;
  }
}

// ---------------- Kernel 4: out_f32 = P16 @ V16, k-loop cut at causal boundary -----
// A tile via async global->LDS DMA; B tile needs a transpose so it stays staged.
__global__ __launch_bounds__(256) void gemm_pv_kernel(
    const _Float16* __restrict__ P, const _Float16* __restrict__ V,
    float* __restrict__ Out, int Ndim)
{
  __shared__ _Float16 As[2][BM * LDT];
  __shared__ _Float16 Bt[2][BN * LDT];
  const int t    = threadIdx.x;
  const int lane = t & 31;
  const int wid  = t >> 5;
  const int r0   = blockIdx.y * BM;
  const int c0   = blockIdx.x * BN;
  const int wm   = (wid >> 2) * 64;
  const int wn   = (wid & 3) * 32;
  const int kend = r0 + BM;                   // P is zero beyond the causal boundary

  const int ar = t >> 1, ak = (t & 1) * 16;
  const int bk = t >> 3, bn = (t & 7) * 16;

  const _Float16* pa = P + (size_t)(r0 + ar) * 4096 + ak;
  const _Float16* pb = V + (size_t)bk * Ndim + c0 + bn;
  const size_t bstep = (size_t)BK * Ndim;

  const uint32_t aoff[2] = { lds_off(&As[0][ar * LDT + ak]), lds_off(&As[1][ar * LDT + ak]) };

  async_copy_b128(aoff[0],      pa);
  async_copy_b128(aoff[0] + 16, pa + 8);
  v8h vb[2];
  vb[0] = *(const v8h*)pb; vb[1] = *(const v8h*)(pb + 8);
  {
    #pragma unroll
    for (int j = 0; j < 8; ++j) {
      Bt[0][(bn + j)     * LDT + bk] = vb[0][j];
      Bt[0][(bn + j + 8) * LDT + bk] = vb[1][j];
    }
  }

  v8f acc[4][2] = {};
  int cur = 0;
  for (int k0 = 0; k0 < kend; k0 += BK) {
    wait_async0();
    __syncthreads();
    const bool more = (k0 + BK) < kend;
    if (more) {
      pa += BK; pb += bstep;
      const int nxt = cur ^ 1;
      async_copy_b128(aoff[nxt],      pa);
      async_copy_b128(aoff[nxt] + 16, pa + 8);
      vb[0] = *(const v8h*)pb; vb[1] = *(const v8h*)(pb + 8);
      if ((k0 + 2*BK) < kend) {
        __builtin_prefetch(pa + BK, 0, 1);
        __builtin_prefetch(pb + bstep, 0, 1);
      }
    }
    mma_step(As[cur], Bt[cur], wm, wn, lane, acc);
    if (more) {
      const int nxt = cur ^ 1;
      #pragma unroll
      for (int j = 0; j < 8; ++j) {
        Bt[nxt][(bn + j)     * LDT + bk] = vb[0][j];
        Bt[nxt][(bn + j + 8) * LDT + bk] = vb[1][j];
      }
    }
    cur ^= 1;
  }

  #pragma unroll
  for (int fm = 0; fm < 4; ++fm)
    #pragma unroll
    for (int fn = 0; fn < 2; ++fn) {
      int n  = c0 + wn + fn * 16 + (lane & 15);
      int mb = r0 + wm + fm * 16 + (lane >> 4) * 8;
      #pragma unroll
      for (int i = 0; i < 8; ++i)
        Out[(size_t)(mb + i) * Ndim + n] = acc[fm][fn][i];
    }
}

extern "C" void kernel_launch(void* const* d_in, const int* in_sizes, int n_in,
                              void* d_out, int out_size, void* d_ws, size_t ws_size,
                              hipStream_t stream) {
  (void)in_sizes; (void)n_in; (void)out_size; (void)ws_size;
  const float* X  = (const float*)d_in[0];
  const float* WQ = (const float*)d_in[1];
  const float* WK = (const float*)d_in[2];
  const float* WV = (const float*)d_in[3];
  float* out = (float*)d_out;

  char* ws = (char*)d_ws;
  const size_t SZ_QKV = (size_t)4096 * 2048 * sizeof(_Float16); // 16 MB each
  _Float16* Q16 = (_Float16*)(ws);
  _Float16* K16 = (_Float16*)(ws + SZ_QKV);
  _Float16* V16 = (_Float16*)(ws + 2 * SZ_QKV);
  _Float16* S16 = (_Float16*)(ws + 3 * SZ_QKV);               // 4096x4096 f16, P in place

  dim3 blk(256);
  dim3 gProj(2048 / BN, 4096 / BM);
  gemm_xw_kernel<<<gProj, blk, 0, stream>>>(X, WQ, Q16, 2048, 2048);
  gemm_xw_kernel<<<gProj, blk, 0, stream>>>(X, WK, K16, 2048, 2048);
  gemm_xw_kernel<<<gProj, blk, 0, stream>>>(X, WV, V16, 2048, 2048);

  dim3 gS(4096 / BN, 4096 / BM);
  gemm_qk_kernel<<<gS, blk, 0, stream>>>(Q16, K16, S16, 2048, 4096);

  softmax_kernel<<<4096, blk, 0, stream>>>(S16);

  dim3 gO(2048 / BN, 4096 / BM);
  gemm_pv_kernel<<<gO, blk, 0, stream>>>(S16, V16, out, 2048);
}
